// OuterProductMean_45990509805638
// MI455X (gfx1250) — matrix operand order repaired and checked by reading, not compiled, not run on hardware
//
#include <hip/hip_runtime.h>

#define DIM   384
#define MROWS 512
#define NROWS 512
#define PDIM  128

typedef __attribute__((ext_vector_type(16))) _Float16 v16h;
typedef __attribute__((ext_vector_type(8)))  _Float16 v8h;
typedef __attribute__((ext_vector_type(2)))  _Float16 h2;
typedef __attribute__((ext_vector_type(8)))  float    v8f;

__device__ __forceinline__ float waveReduceSum(float v) {
#pragma unroll
    for (int off = 16; off > 0; off >>= 1)
        v += __shfl_xor(v, off, 32);
    return v;
}

// One wave per row: LayerNorm(dim=384) then Linear(384 -> 16), output f16.
// Rows 0..511 -> x -> xh ; rows 512..1023 -> y -> yh.
__global__ void reduce_ln_kernel(const float* __restrict__ x,
                                 const float* __restrict__ y,
                                 const float* __restrict__ gamma,
                                 const float* __restrict__ beta,
                                 const float* __restrict__ w_red,
                                 const float* __restrict__ b_red,
                                 _Float16* __restrict__ xh,
                                 _Float16* __restrict__ yh) {
    const int row  = blockIdx.x * (blockDim.x >> 5) + (threadIdx.x >> 5);
    const int lane = threadIdx.x & 31;
    const float* src = (row < MROWS) ? (x + (size_t)row * DIM)
                                     : (y + (size_t)(row - MROWS) * DIM);
    float s1 = 0.f, s2 = 0.f;
    for (int d = lane; d < DIM; d += 32) {
        float v = src[d];
        s1 += v; s2 += v * v;
    }
    s1 = waveReduceSum(s1);
    s2 = waveReduceSum(s2);
    const float mu   = s1 * (1.0f / DIM);
    const float var  = s2 * (1.0f / DIM) - mu * mu;
    const float rinv = rsqrtf(var + 1e-5f);
    if (lane < 16) {
        float acc = b_red[lane];
        for (int d = 0; d < DIM; ++d) {
            float tn = (src[d] - mu) * rinv * gamma[d] + beta[d];
            acc += tn * w_red[d * 16 + lane];
        }
        _Float16* dst = (row < MROWS) ? (xh + row * 16) : (yh + (row - MROWS) * 16);
        dst[lane] = (_Float16)acc;
    }
}

// Pack w_out[256,128] (f32) into per-lane f16 B fragments for
// v_wmma_f32_16x16x32_f16. B (32x16 KxN): lane holds column N = lane%16;
// element e holds K = 8*(lane/16) + (e<8 ? e : 8+e). Layout: [pt][kb][lane][16].
__global__ void pack_bfrag_kernel(const float* __restrict__ w_out,
                                  _Float16* __restrict__ bfrag) {
    const int t = blockIdx.x * blockDim.x + threadIdx.x;   // 0..2047
    if (t >= 8 * 8 * 32) return;
    const int lane = t & 31;
    const int kb   = (t >> 5) & 7;
    const int pt   = t >> 8;
    const int p     = pt * 16 + (lane & 15);
    const int kbase = kb * 32 + ((lane < 16) ? 0 : 8);
    _Float16* dst = bfrag + (size_t)t * 16;
#pragma unroll
    for (int e = 0; e < 16; ++e)
        dst[e] = (_Float16)w_out[(size_t)(kbase + e + (e >= 8 ? 8 : 0)) * PDIM + p];
}

// Main kernel: out[m,n,p] = sum_k A[(m,n),k] * w_out[k,p] + b_out[p],
// A[(m,n),16i+j] = xh[m,i]*yh[n,j] built in registers with packed f16 muls.
// Block = 256 threads = 8 waves; wave w owns p-tile w (16 p columns) and
// keeps its 8 B fragments resident. Each tile = (fixed m, 16 n rows).
__global__ void __launch_bounds__(256)
opm_wmma_kernel(const _Float16* __restrict__ xh,
                const _Float16* __restrict__ yh,
                const _Float16* __restrict__ bfrag,
                const float* __restrict__ b_out,
                float* __restrict__ out) {
    const int lane = threadIdx.x & 31;
    const int pt   = threadIdx.x >> 5;           // 0..7 : p-tile
    const int row  = lane & 15;                  // n-row within tile / p col
    const int j0   = (lane < 16) ? 0 : 8;        // yh j-slice per lane half

    // Resident B fragments (64 VGPRs), constant across all output tiles.
    v16h bf[8];
    const v16h* bsrc = (const v16h*)bfrag;
#pragma unroll
    for (int kb = 0; kb < 8; ++kb)
        bf[kb] = bsrc[(pt * 8 + kb) * 32 + lane];

    const float bias = b_out[pt * 16 + row];

    const int nTiles = MROWS * (NROWS / 16);     // 16384
    for (int t = blockIdx.x; t < nTiles; t += gridDim.x) {
        const int m  = t >> 5;                   // t / 32
        const int n0 = (t & 31) << 4;            // n-tile base

        // 8 consecutive yh halves per lane (one 16B load)
        const v8h yv = *(const v8h*)(yh + (size_t)(n0 + row) * 16 + j0);
        const h2* yv2 = (const h2*)&yv;          // 4 packed pairs

        // 16 wave-uniform xh scalars for this m (32B, scalar-loadable)
        const _Float16* xp = xh + m * 16;
        _Float16 xv[16];
#pragma unroll
        for (int i = 0; i < 16; ++i) xv[i] = xp[i];

        v8f acc = {0.f, 0.f, 0.f, 0.f, 0.f, 0.f, 0.f, 0.f};
#pragma unroll
        for (int kb = 0; kb < 8; ++kb) {
            // A frag, ISA 16-bit A 16x32 layout:
            //  lanes 0-15 : e0-7 -> K=0..7   (i=2kb),  e8-15 -> K=16..23 (i=2kb+1)
            //  lanes16-31 : e0-7 -> K=8..15  (i=2kb),  e8-15 -> K=24..31 (i=2kb+1)
            const h2 xa2 = { xv[2 * kb],     xv[2 * kb]     };
            const h2 xb2 = { xv[2 * kb + 1], xv[2 * kb + 1] };
            v16h a;
            h2* a2 = (h2*)&a;
#pragma unroll
            for (int e2 = 0; e2 < 4; ++e2) {
                a2[e2]     = xa2 * yv2[e2];      // v_pk_mul_f16
                a2[e2 + 4] = xb2 * yv2[e2];      // v_pk_mul_f16
            }
            acc = __builtin_amdgcn_wmma_f32_16x16x32_f16(
                false, a, false, bf[kb], (short)0, acc, false, false);
        }

        // C/D layout: VGPR g, lanes 0-15 -> n-row g, lanes 16-31 -> n-row 8+g
        float* op = out + ((size_t)m * NROWS + n0 + ((lane >> 4) << 3)) * PDIM
                        + pt * 16 + row;
#pragma unroll
        for (int g = 0; g < 8; ++g)
            op[(size_t)g * PDIM] = acc[g] + bias;
    }
}

extern "C" void kernel_launch(void* const* d_in, const int* in_sizes, int n_in,
                              void* d_out, int out_size, void* d_ws, size_t ws_size,
                              hipStream_t stream) {
    const float* x     = (const float*)d_in[0];
    const float* y     = (const float*)d_in[1];
    const float* gamma = (const float*)d_in[2];
    const float* beta  = (const float*)d_in[3];
    const float* w_red = (const float*)d_in[4];
    const float* b_red = (const float*)d_in[5];
    const float* w_out = (const float*)d_in[6];
    const float* b_out = (const float*)d_in[7];
    float* out = (float*)d_out;

    _Float16* xh    = (_Float16*)d_ws;                    // 512*16 f16
    _Float16* yh    = xh + MROWS * 16;                    // 512*16 f16
    _Float16* bfrag = yh + NROWS * 16;                    // 8*8*32*16 f16 = 64KB

    // 1024 rows, one wave each, 4 waves per block
    reduce_ln_kernel<<<256, 128, 0, stream>>>(x, y, gamma, beta, w_red, b_red, xh, yh);
    // 2048 fragment lanes
    pack_bfrag_kernel<<<8, 256, 0, stream>>>(w_out, bfrag);
    // 16384 output tiles over 512 blocks (32 tiles/block, B frags amortized)
    opm_wmma_kernel<<<512, 256, 0, stream>>>(xh, yh, bfrag, b_out, out);
}
